// PotNet_60833916780661
// MI455X (gfx1250) — matrix operand order repaired and checked by reading, not compiled, not run on hardware
//
#include <hip/hip_runtime.h>

#define N_NODES 20000
#define NE      320000
#define FD      256
#define K1      768        // 3*F
#define EPS     1e-5f

typedef __attribute__((ext_vector_type(16))) __bf16 v16bf;
typedef __attribute__((ext_vector_type(8)))  float  v8f;

typedef unsigned short ushort_t;

__device__ __forceinline__ ushort_t f2bf(float f) {
    unsigned int u = __float_as_uint(f);
    u += 0x7FFFu + ((u >> 16) & 1u);      // round-to-nearest-even
    return (ushort_t)(u >> 16);
}

__device__ __forceinline__ float silu_f(float x) {
    return x / (1.0f + __expf(-x));
}

__device__ __forceinline__ void atomAddF(float* p, float v) {
    __hip_atomic_fetch_add(p, v, __ATOMIC_RELAXED, __HIP_MEMORY_SCOPE_AGENT);
}

union Frag32B { uint4 u[2]; v16bf v; };

// ---------------------------------------------------------------------------
// Pack weights transposed to [N][K] bf16 so WMMA B-fragments are contiguous.
// ---------------------------------------------------------------------------
__global__ void pack_w(const float* __restrict__ W1f, const float* __restrict__ W1m,
                       const float* __restrict__ W2f, const float* __restrict__ W2m,
                       ushort_t* __restrict__ w1fT, ushort_t* __restrict__ w1mT,
                       ushort_t* __restrict__ w2fT, ushort_t* __restrict__ w2mT) {
    const int S1 = K1 * FD;   // 196608
    const int S2 = FD * FD;   // 65536
    int t = blockIdx.x * 256 + threadIdx.x;
    if (t < S1) {
        int n = t / K1, k = t - n * K1;
        w1fT[t] = f2bf(W1f[(size_t)k * FD + n]);
        w1mT[t] = f2bf(W1m[(size_t)k * FD + n]);
    } else {
        int u = t - S1;
        if (u < S2) {
            int n = u / FD, k = u - n * FD;
            w2fT[u] = f2bf(W2f[(size_t)k * FD + n]);
            w2mT[u] = f2bf(W2m[(size_t)k * FD + n]);
        }
    }
}

__global__ void zero_k(float* __restrict__ p, int count) {
    int t = blockIdx.x * 256 + threadIdx.x;
    if (t < count) p[t] = 0.0f;
}

// ---------------------------------------------------------------------------
// Edge MLP: per block, 32 edges. 4 waves; each wave owns 64 output columns
// and BOTH 16-row M-tiles (so every B fragment feeds 2 WMMAs per branch).
// Layer1: [32 x 768] @ [768 x 256] (x2 branches), SiLU -> LDS bf16
// Layer2: [32 x 256] @ [256 x 256] (x2 branches) -> t2f/t2m + BN stats
// LDS: h tile (49.7 KB) is dead after layer-1 K-loop; s1f/s1m alias it.
// ---------------------------------------------------------------------------
__global__ __launch_bounds__(128) void edge_mlp(
    const float* __restrict__ x, const int* __restrict__ ei,
    const float* __restrict__ ea,
    const ushort_t* __restrict__ w1fT, const float* __restrict__ b1f,
    const ushort_t* __restrict__ w2fT, const float* __restrict__ b2f,
    const ushort_t* __restrict__ w1mT, const float* __restrict__ b1m,
    const ushort_t* __restrict__ w2mT, const float* __restrict__ b2m,
    float* __restrict__ t2f, float* __restrict__ t2m,
    float* __restrict__ sumF, float* __restrict__ sumsqF)
{
    constexpr int HP = K1 + 8;   // 776 ushorts/row: bank-conflict-free A loads
    constexpr int TP = FD + 8;   // 264

    __shared__ ushort_t smem[32 * HP];   // 49.66 KB, reused for s1f/s1m
    __shared__ int sdst[32], ssrc[32];

    ushort_t* hs  = smem;                // [32][HP]   h tile (bf16)
    ushort_t* s1f = smem;                // [32][TP]   aliases hs after barrier
    ushort_t* s1m = smem + 32 * TP;      // [32][TP]

    const int tid   = threadIdx.x;
    const int lane  = tid & 31;
    const int wv    = tid >> 5;          // 0..3 -> column block of 64
    const int e0    = blockIdx.x * 32;
    const int mrow  = lane & 15;
    const int khalf = (lane >> 4) & 1;

    if (tid < 32) {
        ssrc[tid] = ei[e0 + tid];        // edge_index[0] = src
        sdst[tid] = ei[NE + e0 + tid];   // edge_index[1] = dst
    }
    __syncthreads();

    // Gather h = [x[dst] | x[src] | edge_attr] into LDS as bf16.
    // 32 rows * 192 float4 = 6144 float4 / 128 threads = 48 iters.
    for (int it = 0; it < 48; ++it) {
        int q  = it * 128 + tid;
        int m  = q / 192;
        int c4 = q - m * 192;
        const float* p;
        if (c4 < 64)       p = x  + (size_t)sdst[m] * FD + c4 * 4;
        else if (c4 < 128) p = x  + (size_t)ssrc[m] * FD + (c4 - 64) * 4;
        else               p = ea + (size_t)(e0 + m) * FD + (c4 - 128) * 4;
        float4 v = *(const float4*)p;
        uint2 pk;
        pk.x = (unsigned)f2bf(v.x) | ((unsigned)f2bf(v.y) << 16);
        pk.y = (unsigned)f2bf(v.z) | ((unsigned)f2bf(v.w) << 16);
        *(uint2*)(&hs[m * HP + c4 * 4]) = pk;
    }
    __syncthreads();

    v8f accF[2][4], accM[2][4];
    const v8f vzero = {};
#pragma unroll
    for (int mi = 0; mi < 2; ++mi)
#pragma unroll
        for (int j = 0; j < 4; ++j) { accF[mi][j] = vzero; accM[mi][j] = vzero; }

    // ---------------- Layer 1: K = 768, 24 WMMA K-steps ----------------
    for (int kt = 0; kt < 24; ++kt) {
        const int k0 = kt * 32;
        Frag32B a0, a1;
        const ushort_t* ap0 = &hs[mrow * HP + k0 + khalf * 8];
        a0.u[0] = *(const uint4*)(ap0);
        a0.u[1] = *(const uint4*)(ap0 + 16);
        const ushort_t* ap1 = &hs[(16 + mrow) * HP + k0 + khalf * 8];
        a1.u[0] = *(const uint4*)(ap1);
        a1.u[1] = *(const uint4*)(ap1 + 16);
        const int kb = k0 + khalf * 8;
#pragma unroll
        for (int j = 0; j < 4; ++j) {
            const int n = wv * 64 + j * 16 + mrow;
            Frag32B bf, bm;
            const ushort_t* bp = w1fT + (size_t)n * K1 + kb;
            bf.u[0] = *(const uint4*)(bp);
            bf.u[1] = *(const uint4*)(bp + 16);
            const ushort_t* mp = w1mT + (size_t)n * K1 + kb;
            bm.u[0] = *(const uint4*)(mp);
            bm.u[1] = *(const uint4*)(mp + 16);
            accF[0][j] = __builtin_amdgcn_wmma_f32_16x16x32_bf16(
                false, a0.v, false, bf.v, (short)0, accF[0][j], false, false);
            accF[1][j] = __builtin_amdgcn_wmma_f32_16x16x32_bf16(
                false, a1.v, false, bf.v, (short)0, accF[1][j], false, false);
            accM[0][j] = __builtin_amdgcn_wmma_f32_16x16x32_bf16(
                false, a0.v, false, bm.v, (short)0, accM[0][j], false, false);
            accM[1][j] = __builtin_amdgcn_wmma_f32_16x16x32_bf16(
                false, a1.v, false, bm.v, (short)0, accM[1][j], false, false);
        }
    }

    // All waves done reading hs before s1f/s1m overwrite it.
    __syncthreads();

    // Layer-1 epilogue: bias + SiLU -> bf16 LDS
#pragma unroll
    for (int j = 0; j < 4; ++j) {
        const int n = wv * 64 + j * 16 + mrow;
        const float bfv = b1f[n], bmv = b1m[n];
#pragma unroll
        for (int mi = 0; mi < 2; ++mi) {
#pragma unroll
            for (int r = 0; r < 8; ++r) {
                const int m = mi * 16 + r + khalf * 8;
                s1f[m * TP + n] = f2bf(silu_f(accF[mi][j][r] + bfv));
                s1m[m * TP + n] = f2bf(silu_f(accM[mi][j][r] + bmv));
            }
            accF[mi][j] = vzero; accM[mi][j] = vzero;
        }
    }
    __syncthreads();

    // ---------------- Layer 2: K = 256, 8 WMMA K-steps ----------------
    for (int kt = 0; kt < 8; ++kt) {
        const int k0 = kt * 32;
        Frag32B af0, af1, am0, am1;
        const ushort_t* p0 = &s1f[mrow * TP + k0 + khalf * 8];
        af0.u[0] = *(const uint4*)(p0);
        af0.u[1] = *(const uint4*)(p0 + 16);
        const ushort_t* p1 = &s1f[(16 + mrow) * TP + k0 + khalf * 8];
        af1.u[0] = *(const uint4*)(p1);
        af1.u[1] = *(const uint4*)(p1 + 16);
        const ushort_t* q0 = &s1m[mrow * TP + k0 + khalf * 8];
        am0.u[0] = *(const uint4*)(q0);
        am0.u[1] = *(const uint4*)(q0 + 16);
        const ushort_t* q1 = &s1m[(16 + mrow) * TP + k0 + khalf * 8];
        am1.u[0] = *(const uint4*)(q1);
        am1.u[1] = *(const uint4*)(q1 + 16);
        const int kb = k0 + khalf * 8;
#pragma unroll
        for (int j = 0; j < 4; ++j) {
            const int n = wv * 64 + j * 16 + mrow;
            Frag32B bf, bm;
            const ushort_t* bp = w2fT + (size_t)n * FD + kb;
            bf.u[0] = *(const uint4*)(bp);
            bf.u[1] = *(const uint4*)(bp + 16);
            const ushort_t* mp = w2mT + (size_t)n * FD + kb;
            bm.u[0] = *(const uint4*)(mp);
            bm.u[1] = *(const uint4*)(mp + 16);
            accF[0][j] = __builtin_amdgcn_wmma_f32_16x16x32_bf16(
                false, af0.v, false, bf.v, (short)0, accF[0][j], false, false);
            accF[1][j] = __builtin_amdgcn_wmma_f32_16x16x32_bf16(
                false, af1.v, false, bf.v, (short)0, accF[1][j], false, false);
            accM[0][j] = __builtin_amdgcn_wmma_f32_16x16x32_bf16(
                false, am0.v, false, bm.v, (short)0, accM[0][j], false, false);
            accM[1][j] = __builtin_amdgcn_wmma_f32_16x16x32_bf16(
                false, am1.v, false, bm.v, (short)0, accM[1][j], false, false);
        }
    }

    // Final epilogue: bias, store t2f/t2m fp32, accumulate edge-BN stats.
#pragma unroll
    for (int j = 0; j < 4; ++j) {
        const int n = wv * 64 + j * 16 + mrow;
        const float b2fv = b2f[n], b2mv = b2m[n];
        float ps = 0.0f, psq = 0.0f;
#pragma unroll
        for (int mi = 0; mi < 2; ++mi) {
#pragma unroll
            for (int r = 0; r < 8; ++r) {
                const int m = mi * 16 + r + khalf * 8;
                const float vf = accF[mi][j][r] + b2fv;
                const float vm = accM[mi][j][r] + b2mv;
                const size_t o = (size_t)(e0 + m) * FD + n;
                t2f[o] = vf;
                t2m[o] = vm;
                ps  += vf;
                psq += vf * vf;
            }
        }
        atomAddF(&sumF[n],   ps);
        atomAddF(&sumsqF[n], psq);
    }
}

// ---------------------------------------------------------------------------
// gate = sigmoid(BN(t2f)); msg = gate * t2m; scatter-add into agg[dst]
// ---------------------------------------------------------------------------
__global__ void msg_scatter(const float* __restrict__ t2f, const float* __restrict__ t2m,
                            const int* __restrict__ ei,
                            const float* __restrict__ sumF, const float* __restrict__ sumsqF,
                            const float* __restrict__ g, const float* __restrict__ beta,
                            float* __restrict__ agg)
{
    int t = blockIdx.x * 256 + threadIdx.x;      // one thread per 4 features
    int e = t >> 6;                              // FD/4 = 64 threads per edge
    if (e >= NE) return;
    int f = (t & 63) * 4;
    int d = ei[NE + e];

    const float invE = 1.0f / (float)NE;
    float4 tf = *(const float4*)(t2f + (size_t)e * FD + f);
    float4 tm = *(const float4*)(t2m + (size_t)e * FD + f);
    float tfv[4] = {tf.x, tf.y, tf.z, tf.w};
    float tmv[4] = {tm.x, tm.y, tm.z, tm.w};
    float* ag = agg + (size_t)d * FD + f;

#pragma unroll
    for (int i = 0; i < 4; ++i) {
        float mu  = sumF[f + i] * invE;
        float var = sumsqF[f + i] * invE - mu * mu;
        float z   = (tfv[i] - mu) * __frsqrt_rn(var + EPS) * g[f + i] + beta[f + i];
        float gate = 1.0f / (1.0f + __expf(-z));
        atomAddF(&ag[i], gate * tmv[i]);
    }
}

// Column stats of agg [N, F] for the node BatchNorm.
__global__ void col_stats(const float* __restrict__ agg,
                          float* __restrict__ sumA, float* __restrict__ sumsqA) {
    int f  = threadIdx.x;
    int r0 = blockIdx.x * 64;
    float s = 0.0f, sq = 0.0f;
    for (int r = 0; r < 64; ++r) {
        int n = r0 + r;
        if (n < N_NODES) {
            float v = agg[(size_t)n * FD + f];
            s += v; sq += v * v;
        }
    }
    atomAddF(&sumA[f],   s);
    atomAddF(&sumsqA[f], sq);
}

// out = relu(x + BN(agg))
__global__ void final_k(const float* __restrict__ x, const float* __restrict__ agg,
                        const float* __restrict__ sumA, const float* __restrict__ sumsqA,
                        const float* __restrict__ g, const float* __restrict__ beta,
                        float* __restrict__ out) {
    int t = blockIdx.x * 256 + threadIdx.x;
    if (t >= N_NODES * FD) return;
    int f = t & (FD - 1);
    const float invN = 1.0f / (float)N_NODES;
    float mu  = sumA[f] * invN;
    float var = sumsqA[f] * invN - mu * mu;
    float v = x[t] + (agg[t] - mu) * __frsqrt_rn(var + EPS) * g[f] + beta[f];
    out[t] = fmaxf(v, 0.0f);
}

// ---------------------------------------------------------------------------
extern "C" void kernel_launch(void* const* d_in, const int* in_sizes, int n_in,
                              void* d_out, int out_size, void* d_ws, size_t ws_size,
                              hipStream_t stream) {
    const float* x        = (const float*)d_in[0];
    const int*   ei       = (const int*)  d_in[1];
    const float* ea       = (const float*)d_in[2];
    const float* W1f      = (const float*)d_in[3];
    const float* b1f      = (const float*)d_in[4];
    const float* W2f      = (const float*)d_in[5];
    const float* b2f      = (const float*)d_in[6];
    const float* W1m      = (const float*)d_in[7];
    const float* b1m      = (const float*)d_in[8];
    const float* W2m      = (const float*)d_in[9];
    const float* b2m      = (const float*)d_in[10];
    const float* g_int    = (const float*)d_in[11];
    const float* beta_int = (const float*)d_in[12];
    const float* g_bn     = (const float*)d_in[13];
    const float* beta_bn  = (const float*)d_in[14];
    float* out = (float*)d_out;

    // Workspace carve-up
    char* ws = (char*)d_ws;
    float* t2f = (float*)ws;  ws += sizeof(float) * (size_t)NE * FD;
    float* t2m = (float*)ws;  ws += sizeof(float) * (size_t)NE * FD;
    float* agg = (float*)ws;  ws += sizeof(float) * (size_t)N_NODES * FD;
    float* sumF   = (float*)ws; ws += sizeof(float) * FD;
    float* sumsqF = (float*)ws; ws += sizeof(float) * FD;
    float* sumA   = (float*)ws; ws += sizeof(float) * FD;
    float* sumsqA = (float*)ws; ws += sizeof(float) * FD;
    ushort_t* w1fT = (ushort_t*)ws; ws += sizeof(ushort_t) * (size_t)FD * K1;
    ushort_t* w1mT = (ushort_t*)ws; ws += sizeof(ushort_t) * (size_t)FD * K1;
    ushort_t* w2fT = (ushort_t*)ws; ws += sizeof(ushort_t) * (size_t)FD * FD;
    ushort_t* w2mT = (ushort_t*)ws; ws += sizeof(ushort_t) * (size_t)FD * FD;

    // 1. pack weights (bf16, transposed)
    pack_w<<<(K1 * FD + FD * FD + 255) / 256, 256, 0, stream>>>(
        W1f, W1m, W2f, W2m, w1fT, w1mT, w2fT, w2mT);

    // 2. zero agg + 4 stat arrays (they are contiguous after agg)
    const int zcount = N_NODES * FD + 4 * FD;
    zero_k<<<(zcount + 255) / 256, 256, 0, stream>>>(agg, zcount);

    // 3. edge MLP (WMMA) — 32 edges / block
    edge_mlp<<<NE / 32, 128, 0, stream>>>(
        x, ei, ea, w1fT, b1f, w2fT, b2f, w1mT, b1m, w2mT, b2m,
        t2f, t2m, sumF, sumsqF);

    // 4. gate + message + scatter-add
    msg_scatter<<<(NE * (FD / 4) + 255) / 256, 256, 0, stream>>>(
        t2f, t2m, ei, sumF, sumsqF, g_int, beta_int, agg);

    // 5. node BN stats
    col_stats<<<(N_NODES + 63) / 64, 256, 0, stream>>>(agg, sumA, sumsqA);

    // 6. residual + BN + relu
    final_k<<<(N_NODES * FD + 255) / 256, 256, 0, stream>>>(
        x, agg, sumA, sumsqA, g_bn, beta_bn, out);
}
